// FocusPolicy_86002425135758
// MI455X (gfx1250) — matrix operand order repaired and checked by reading, not compiled, not run on hardware
//
#include <hip/hip_runtime.h>
#include <cmath>

typedef __attribute__((ext_vector_type(16))) _Float16 v16h;
typedef __attribute__((ext_vector_type(8)))  float    v8f;
typedef _Float16 half_t;

static inline int cdiv_h(long long a, long long b) { return (int)((a + b - 1) / b); }

#define RPOOL  23
#define RRATIO 2
#define RSCALE 0.25f

template <int N>
__device__ __forceinline__ void wait_async_le()
{
#if __has_builtin(__builtin_amdgcn_s_wait_asynccnt)
  __builtin_amdgcn_s_wait_asynccnt(N);   // N is an integer constant expression
#else
  if (N == 0) asm volatile("s_wait_asynccnt 0x0" ::: "memory");
  else        asm volatile("s_wait_asynccnt 0x2" ::: "memory");
#endif
}

// ===================== misc =====================
__global__ void zero_kernel(float* __restrict__ p, int n)
{
  int i = blockIdx.x * blockDim.x + threadIdx.x;
  if (i < n) p[i] = 0.0f;
}

// ===================== weight packing: [N][K] f32 -> [KPAD][N] f16 (K zero-padded) ==
__global__ void pack_w_kernel(const float* __restrict__ w, half_t* __restrict__ wp,
                              int N, int K, int KPAD)
{
  long long idx = (long long)blockIdx.x * blockDim.x + threadIdx.x;
  if (idx >= (long long)KPAD * N) return;
  int n = (int)(idx % N);
  int k = (int)(idx / N);
  wp[idx] = (k < K) ? (half_t)w[(long long)n * K + k] : (half_t)0.0f;
}

// ===================== implicit-GEMM conv via WMMA =================================
// Block tile: (16*MTILE) x NOUT.  NOUT/16 waves; wave w owns columns [16w,16w+16).
// Per k-step each wave issues MTILE WMMAs reusing one B fragment.
// A tile staged in WMMA fragment order:
//   lane L holds row M=L&15; half h holds K=(h&7)+16*(h>=8)+8*(L>=16).
// B tile (32 x NOUT, [k][n] row-major, contiguous in the packed weights) is staged
// with GLOBAL_LOAD_ASYNC_TO_LDS_B128 (ASYNCcnt), double-buffered: the tile for
// step ks+1 is prefetched while step ks computes; s_wait_asynccnt 2 retires the
// older transfers while the newest 2 per wave stay in flight.
template <int CIN, int HIN, int WIN, int KH, int KW, int STRIDE, int PAD,
          int HOUT, int WOUT, int NOUT, int MTILE, bool IN_NHWC,
          typename InT, typename OutT, bool RELU, bool BIAS>
__global__ __launch_bounds__(256)
void conv_wmma_kernel(const InT* __restrict__ in, const half_t* __restrict__ Wp,
                      const float* __restrict__ bias, OutT* __restrict__ out, int Bn)
{
  constexpr int K        = CIN * KH * KW;
  constexpr int KPAD     = (K + 31) & ~31;
  constexpr int KSTEP    = KPAD / 32;
  constexpr int MROWS    = 16 * MTILE;
  constexpr int NTHREADS = 2 * NOUT;          // launch uses 32*(NOUT/16) threads
  constexpr int NXFER    = (32 * NOUT) / 8;   // b128 transfers per B tile
  __shared__ alignas(32) half_t ldsA[MROWS * 32];
  __shared__ alignas(32) half_t ldsB[2][32 * NOUT];

  const int tid  = threadIdx.x;
  const int lane = tid & 31;
  const int wave = tid >> 5;
  const int m0   = blockIdx.x * MROWS;
  const int M    = Bn * HOUT * WOUT;

  auto issueB = [&](int ks, int buf) {
    const half_t* gsrc = Wp + (long long)ks * (32 * NOUT);
    for (int idx = tid; idx < NXFER; idx += NTHREADS) {
      unsigned lds_off = (unsigned)(unsigned long long)(const void*)&ldsB[buf][idx * 8];
      unsigned goff    = (unsigned)(idx * 16);
      asm volatile("global_load_async_to_lds_b128 %0, %1, %2"
                   :: "v"(lds_off), "v"(goff), "s"(gsrc) : "memory");
    }
  };

  v8f acc[MTILE] = {};
  issueB(0, 0);
  for (int ks = 0; ks < KSTEP; ++ks) {
    const int cur = ks & 1;
    const bool more = (ks + 1) < KSTEP;
    if (more) issueB(ks + 1, cur ^ 1);   // prefetch next B tile into alternate buffer
    // ---- stage A (im2col) in fragment order (overlaps with async B transfers) ----
    for (int idx = tid; idx < MROWS * 32; idx += NTHREADS) {
      int mg = idx >> 5;           // row within block tile
      int kl = idx & 31;
      int k  = ks * 32 + kl;
      int m  = m0 + mg;
      float v = 0.0f;
      if (k < K && m < M) {
        int b   = m / (HOUT * WOUT);
        int rem = m - b * (HOUT * WOUT);
        int oy  = rem / WOUT;
        int ox  = rem - oy * WOUT;
        int c   = k / (KH * KW);
        int kr  = k - c * (KH * KW);
        int r   = kr / KW;
        int s   = kr - r * KW;
        int iy  = oy * STRIDE - PAD + r;
        int ix  = ox * STRIDE - PAD + s;
        if (iy >= 0 && iy < HIN && ix >= 0 && ix < WIN) {
          long long off = IN_NHWC
              ? (((long long)b * HIN + iy) * WIN + ix) * CIN + c
              : (((long long)b * CIN + c) * HIN + iy) * WIN + ix;
          v = (float)in[off];
        }
      }
      int msub = mg >> 4;
      int ml   = mg & 15;
      int Ld   = ml + 16 * ((kl >> 3) & 1);
      int hd   = (kl & 7) + 8 * (kl >> 4);
      ldsA[msub * 512 + Ld * 16 + hd] = (half_t)v;
    }
    // retire the transfers for the current buffer; newest 2 may stay in flight
    if (more) wait_async_le<2>(); else wait_async_le<0>();
    __syncthreads();
    v16h bfrag;
#pragma unroll
    for (int s = 0; s < 16; ++s)
      bfrag[s] = ldsB[cur][((s & 7) + 8 * (lane >> 4) + 16 * (s >> 3)) * NOUT + wave * 16 + (lane & 15)];
#pragma unroll
    for (int ms = 0; ms < MTILE; ++ms) {
      v16h afrag = *(const v16h*)&ldsA[ms * 512 + lane * 16];
      acc[ms] = __builtin_amdgcn_wmma_f32_16x16x32_f16(false, afrag, false, bfrag,
                                                       (short)0, acc[ms], false, false);
    }
    __syncthreads();   // WAR fence: LDS reads done before next iter's stores/async
  }
  // ---- epilogue: D[m = i + 8*(lane>>4)][n = (lane&15) + 16*wave] ----
  const int   n  = (lane & 15) + wave * 16;
  const float bv = BIAS ? bias[n] : 0.0f;
#pragma unroll
  for (int ms = 0; ms < MTILE; ++ms) {
#pragma unroll
    for (int i = 0; i < 8; ++i) {
      int m = m0 + ms * 16 + i + 8 * (lane >> 4);
      if (m < M) {
        float v = acc[ms][i] + bv;
        if (RELU) v = fmaxf(v, 0.0f);
        int b   = m / (HOUT * WOUT);
        int rem = m - b * (HOUT * WOUT);
        out[((long long)b * NOUT + n) * (HOUT * WOUT) + rem] = (OutT)v;
      }
    }
  }
}

// ===================== batch-norm partial statistics (atomic accumulation) =========
template <typename InT>
__global__ void bn_stats_kernel(const InT* __restrict__ in, float* __restrict__ stats,
                                int C, int HW, int Bn, int split)
{
  __shared__ float ssum[256];
  __shared__ float ssq[256];
  const int c     = blockIdx.x / split;
  const int slice = blockIdx.x % split;
  const long long total = (long long)Bn * HW;
  const long long chunk = (total + split - 1) / split;
  const long long lo = (long long)slice * chunk;
  const long long hi = (lo + chunk < total) ? lo + chunk : total;
  float s = 0.0f, q = 0.0f;
  for (long long i = lo + threadIdx.x; i < hi; i += blockDim.x) {
    int b  = (int)(i / HW);
    int pp = (int)(i - (long long)b * HW);
    float v = (float)in[((long long)b * C + c) * HW + pp];
    s += v; q += v * v;
  }
  ssum[threadIdx.x] = s; ssq[threadIdx.x] = q;
  __syncthreads();
  for (int o = 128; o > 0; o >>= 1) {
    if (threadIdx.x < o) {
      ssum[threadIdx.x] += ssum[threadIdx.x + o];
      ssq[threadIdx.x]  += ssq[threadIdx.x + o];
    }
    __syncthreads();
  }
  if (threadIdx.x == 0) {
    atomicAdd(&stats[2 * c],     ssum[0]);
    atomicAdd(&stats[2 * c + 1], ssq[0]);
  }
}

// ===================== fused BN + ReLU + maxpool (finalizes stats inline) ==========
template <typename InT, int KP, int SP, int PP>
__global__ void bn_relu_pool_kernel(const InT* __restrict__ in, const float* __restrict__ stats,
                                    const float* __restrict__ gamma, const float* __restrict__ beta,
                                    float* __restrict__ out, float invCnt,
                                    int C, int Hin, int Win, int Hout, int Wout, int Bn)
{
  long long idx   = (long long)blockIdx.x * blockDim.x + threadIdx.x;
  long long total = (long long)Bn * C * Hout * Wout;
  if (idx >= total) return;
  int x = (int)(idx % Wout);
  long long t = idx / Wout;
  int y = (int)(t % Hout); t /= Hout;
  int c = (int)(t % C);
  int b = (int)(t / C);
  float mean = stats[2 * c] * invCnt;
  float var  = stats[2 * c + 1] * invCnt - mean * mean;
  float rstd = rsqrtf(var + 1e-5f);
  float gg = gamma[c] * rstd;
  float bb = beta[c] - mean * gg;
  float best = -1e30f;
  for (int r = 0; r < KP; ++r) {
    int iy = y * SP - PP + r;
    if (iy < 0 || iy >= Hin) continue;
    for (int s = 0; s < KP; ++s) {
      int ix = x * SP - PP + s;
      if (ix < 0 || ix >= Win) continue;
      float v = (float)in[(((long long)b * C + c) * Hin + iy) * Win + ix];
      v = fmaxf(v * gg + bb, 0.0f);
      best = fmaxf(best, v);
    }
  }
  out[idx] = best;
}

// ===================== affine grid_sample (rotation) ===============================
__global__ void grid_sample_kernel(const float* __restrict__ f, float* __restrict__ out,
                                   int C, int H, int W, int Bn,
                                   float t00, float t01, float t02,
                                   float t10, float t11, float t12)
{
  long long idx   = (long long)blockIdx.x * blockDim.x + threadIdx.x;
  long long total = (long long)Bn * C * H * W;
  if (idx >= total) return;
  int x = (int)(idx % W);
  long long t = idx / W;
  int y = (int)(t % H); t /= H;
  int c = (int)(t % C);
  int b = (int)(t / C);
  float gx = (2.0f * x + 1.0f) / (float)W - 1.0f;
  float gy = (2.0f * y + 1.0f) / (float)H - 1.0f;
  float u = gx * t00 + gy * t01 + t02;
  float v = gx * t10 + gy * t11 + t12;
  float ix = ((u + 1.0f) * (float)W - 1.0f) * 0.5f;
  float iy = ((v + 1.0f) * (float)H - 1.0f) * 0.5f;
  float x0f = floorf(ix), y0f = floorf(iy);
  int x0 = (int)x0f, y0 = (int)y0f;
  float wa = (x0f + 1.0f - ix) * (y0f + 1.0f - iy);
  float wb = (x0f + 1.0f - ix) * (iy - y0f);
  float wc = (ix - x0f) * (y0f + 1.0f - iy);
  float wd = (ix - x0f) * (iy - y0f);
  const float* img = f + ((long long)b * C + c) * H * W;
  auto g = [&](int yy, int xx) -> float {
    bool valid = (xx >= 0) && (xx < W) && (yy >= 0) && (yy < H);
    int yc = yy < 0 ? 0 : (yy > H - 1 ? H - 1 : yy);
    int xc = xx < 0 ? 0 : (xx > W - 1 ? W - 1 : xx);
    float val = img[yc * W + xc];
    return valid ? val : 0.0f;
  };
  out[idx] = g(y0, x0) * wa + g(y0 + 1, x0) * wb + g(y0, x0 + 1) * wc + g(y0 + 1, x0 + 1) * wd;
}

// ===================== ROI-align into the concatenated (Bn,704,23,23) buffer =======
__global__ void roialign_kernel(const float* __restrict__ feat, const float* __restrict__ box,
                                float* __restrict__ hout,
                                int Bn, int C, int H, int W,
                                int nbox, int boxStart, int chanBase)
{
  long long idx   = (long long)blockIdx.x * blockDim.x + threadIdx.x;
  long long total = (long long)Bn * nbox * C * RPOOL * RPOOL;
  if (idx >= total) return;
  int pw = (int)(idx % RPOOL);
  long long t = idx / RPOOL;
  int ph = (int)(t % RPOOL); t /= RPOOL;
  int c = (int)(t % C); t /= C;
  int j = (int)(t % nbox);
  int b = (int)(t / nbox);
  const float* bx = box + (long long)b * 28 + (long long)(boxStart + j) * 4;
  float x1 = bx[0] * RSCALE, y1 = bx[1] * RSCALE;
  float x2 = bx[2] * RSCALE, y2 = bx[3] * RSCALE;
  float bw = fmaxf(x2 - x1, 1.0f) / (float)RPOOL;
  float bh = fmaxf(y2 - y1, 1.0f) / (float)RPOOL;
  const float* img = feat + ((long long)b * C + c) * H * W;
  float accv = 0.0f;
  for (int sy = 0; sy < RRATIO; ++sy) {
    float posy = (float)ph + ((float)sy + 0.5f) / (float)RRATIO;
    float yv = y1 + posy * bh;
    for (int sx = 0; sx < RRATIO; ++sx) {
      float posx = (float)pw + ((float)sx + 0.5f) / (float)RRATIO;
      float xv = x1 + posx * bw;
      bool empty = (yv < -1.0f) || (yv > (float)H) || (xv < -1.0f) || (xv > (float)W);
      float yq = fminf(fmaxf(yv, 0.0f), (float)(H - 1));
      float xq = fminf(fmaxf(xv, 0.0f), (float)(W - 1));
      float y0f = floorf(yq), x0f = floorf(xq);
      int y0 = (int)y0f, x0 = (int)x0f;
      int y1i = (y0 + 1 < H - 1) ? y0 + 1 : H - 1;
      int x1i = (x0 + 1 < W - 1) ? x0 + 1 : W - 1;
      float ly = yq - y0f, lx = xq - x0f;
      float hy = 1.0f - ly, hx = 1.0f - lx;
      float v = img[y0 * W + x0]  * (hy * hx) + img[y0 * W + x1i]  * (hy * lx)
              + img[y1i * W + x0] * (ly * hx) + img[y1i * W + x1i] * (ly * lx);
      accv += empty ? 0.0f : v;
    }
  }
  accv *= 1.0f / (float)(RRATIO * RRATIO);
  hout[(((long long)b * 704 + (chanBase + j * C + c)) * RPOOL + ph) * RPOOL + pw] = accv;
}

// ===================== head: (16,128) @ (12,128)^T + b, tanh =======================
__global__ void head_kernel(const float* __restrict__ act, const float* __restrict__ hw,
                            const float* __restrict__ hb, float* __restrict__ out)
{
  int t = blockIdx.x * blockDim.x + threadIdx.x;
  if (t >= 16 * 12) return;
  int b = t / 12, j = t % 12;
  float s = hb[j];
  const float* a = act + b * 128;
  const float* w = hw + j * 128;
  for (int k = 0; k < 128; ++k) s += a[k] * w[k];
  out[t] = tanhf(s);
}

// ===================== orchestration ===============================================
extern "C" void kernel_launch(void* const* d_in, const int* in_sizes, int n_in,
                              void* d_out, int out_size, void* d_ws, size_t ws_size,
                              hipStream_t stream)
{
  (void)in_sizes; (void)n_in; (void)out_size; (void)ws_size;
  const float* x       = (const float*)d_in[0];
  const float* box     = (const float*)d_in[1];
  const float* stem_w  = (const float*)d_in[2];
  const float* stem_g  = (const float*)d_in[3];
  const float* stem_bb = (const float*)d_in[4];
  const float* conv1_w = (const float*)d_in[5];
  const float* conv1_b = (const float*)d_in[6];
  const float* bn1_g   = (const float*)d_in[7];
  const float* bn1_b   = (const float*)d_in[8];
  const float* conv2_w = (const float*)d_in[9];
  const float* conv2_b = (const float*)d_in[10];
  const float* fc1_w   = (const float*)d_in[11];
  const float* fc1_b   = (const float*)d_in[12];
  const float* head_w  = (const float*)d_in[13];
  const float* head_b  = (const float*)d_in[14];

  const int Bn = 16;
  char* p = (char*)d_ws;
  auto carve = [&](size_t bytes) -> void* {
    void* r = (void*)p;
    p += (bytes + 255) & ~(size_t)255;
    return r;
  };

  half_t* stem_out = (half_t*)carve((size_t)Bn * 64 * 256 * 256 * sizeof(half_t));
  float*  f        = (float*) carve((size_t)Bn * 64 * 128 * 128 * sizeof(float));
  float*  xt1      = (float*) carve((size_t)Bn * 64 * 128 * 128 * sizeof(float));
  float*  xt2      = (float*) carve((size_t)Bn * 64 * 128 * 128 * sizeof(float));
  float*  hbuf     = (float*) carve((size_t)Bn * 704 * 23 * 23 * sizeof(float));
  float*  c1out    = (float*) carve((size_t)Bn * 64 * 21 * 21 * sizeof(float));
  float*  pool1    = (float*) carve((size_t)Bn * 64 * 10 * 10 * sizeof(float));
  float*  act2048  = (float*) carve((size_t)Bn * 2048 * sizeof(float));
  float*  fc1out   = (float*) carve((size_t)Bn * 128 * sizeof(float));
  float*  stats0   = (float*) carve(128 * sizeof(float));
  float*  stats1   = (float*) carve(128 * sizeof(float));
  half_t* WstemP   = (half_t*)carve((size_t)160 * 64 * sizeof(half_t));
  half_t* Wc1P     = (half_t*)carve((size_t)6336 * 64 * sizeof(half_t));
  half_t* Wc2P     = (half_t*)carve((size_t)576 * 32 * sizeof(half_t));
  half_t* Wfc1P    = (half_t*)carve((size_t)2048 * 128 * sizeof(half_t));

  // --- pack weights to f16 [KPAD][N]; zero stats accumulators ---
  pack_w_kernel<<<cdiv_h(160LL * 64, 256),   256, 0, stream>>>(stem_w,  WstemP, 64, 147, 160);
  pack_w_kernel<<<cdiv_h(6336LL * 64, 256),  256, 0, stream>>>(conv1_w, Wc1P, 64, 6336, 6336);
  pack_w_kernel<<<cdiv_h(576LL * 32, 256),   256, 0, stream>>>(conv2_w, Wc2P, 32, 576, 576);
  pack_w_kernel<<<cdiv_h(2048LL * 128, 256), 256, 0, stream>>>(fc1_w,   Wfc1P, 128, 2048, 2048);
  zero_kernel<<<1, 256, 0, stream>>>(stats0, 128);
  zero_kernel<<<1, 256, 0, stream>>>(stats1, 128);

  // --- stem conv 7x7 s2 p3, NHWC f32 in -> NCHW f16 out (WMMA, 64x64 tiles) ---
  conv_wmma_kernel<3, 512, 512, 7, 7, 2, 3, 256, 256, 64, 4, true, float, half_t, false, false>
      <<<cdiv_h((long long)Bn * 256 * 256, 64), 128, 0, stream>>>(x, WstemP, nullptr, stem_out, Bn);

  // --- BN stats (split + atomics), fused BN+ReLU+maxpool(3,2,1) -> f ---
  bn_stats_kernel<half_t><<<64 * 32, 256, 0, stream>>>(stem_out, stats0, 64, 256 * 256, Bn, 32);
  {
    long long total = (long long)Bn * 64 * 128 * 128;
    bn_relu_pool_kernel<half_t, 3, 2, 1><<<cdiv_h(total, 256), 256, 0, stream>>>(
        stem_out, stats0, stem_g, stem_bb, f, 1.0f / (float)(Bn * 256 * 256),
        64, 256, 256, 128, 128, Bn);
  }

  // --- rotated resamples (theta = [[c, -s, 0],[s, c, 0]], a = -/+5 deg) ---
  {
    const double PI = 3.14159265358979323846;
    double a1 = -5.0 * PI / 180.0, a2 = 5.0 * PI / 180.0;
    long long total = (long long)Bn * 64 * 128 * 128;
    grid_sample_kernel<<<cdiv_h(total, 256), 256, 0, stream>>>(
        f, xt1, 64, 128, 128, Bn,
        (float)std::cos(a1), (float)(-std::sin(a1)), 0.0f,
        (float)std::sin(a1), (float)std::cos(a1), 0.0f);
    grid_sample_kernel<<<cdiv_h(total, 256), 256, 0, stream>>>(
        f, xt2, 64, 128, 128, Bn,
        (float)std::cos(a2), (float)(-std::sin(a2)), 0.0f,
        (float)std::sin(a2), (float)std::cos(a2), 0.0f);
  }

  // --- ROI aligns into concatenated hbuf (Bn, 704, 23, 23) ---
  {
    long long t7 = (long long)Bn * 7 * 64 * RPOOL * RPOOL;
    long long t1 = (long long)Bn * 1 * 64 * RPOOL * RPOOL;
    roialign_kernel<<<cdiv_h(t7, 256), 256, 0, stream>>>(f,   box, hbuf, Bn, 64, 128, 128, 7, 0, 0);
    roialign_kernel<<<cdiv_h(t1, 256), 256, 0, stream>>>(xt1, box, hbuf, Bn, 64, 128, 128, 1, 0, 448);
    roialign_kernel<<<cdiv_h(t1, 256), 256, 0, stream>>>(xt1, box, hbuf, Bn, 64, 128, 128, 1, 1, 512);
    roialign_kernel<<<cdiv_h(t1, 256), 256, 0, stream>>>(xt2, box, hbuf, Bn, 64, 128, 128, 1, 0, 576);
    roialign_kernel<<<cdiv_h(t1, 256), 256, 0, stream>>>(xt2, box, hbuf, Bn, 64, 128, 128, 1, 1, 640);
  }

  // --- conv1 3x3 valid (704 -> 64) + bias (WMMA, K = 6336) ---
  conv_wmma_kernel<704, 23, 23, 3, 3, 1, 0, 21, 21, 64, 4, false, float, float, false, true>
      <<<cdiv_h((long long)Bn * 21 * 21, 64), 128, 0, stream>>>(hbuf, Wc1P, conv1_b, c1out, Bn);

  // --- bn1 stats, BN+ReLU+maxpool(2,2) -> pool1 (Bn,64,10,10) ---
  bn_stats_kernel<float><<<64 * 4, 256, 0, stream>>>(c1out, stats1, 64, 21 * 21, Bn, 4);
  bn_relu_pool_kernel<float, 2, 2, 0><<<cdiv_h((long long)Bn * 64 * 10 * 10, 256), 256, 0, stream>>>(
      c1out, stats1, bn1_g, bn1_b, pool1, 1.0f / (float)(Bn * 21 * 21),
      64, 21, 21, 10, 10, Bn);

  // --- conv2 3x3 valid (64 -> 32) + bias + ReLU -> (Bn, 2048) flat ---
  conv_wmma_kernel<64, 10, 10, 3, 3, 1, 0, 8, 8, 32, 4, false, float, float, true, true>
      <<<cdiv_h((long long)Bn * 8 * 8, 64), 64, 0, stream>>>(pool1, Wc2P, conv2_b, act2048, Bn);

  // --- fc1 (2048 -> 128) + bias + ReLU as WMMA GEMM (M=16, one block) ---
  conv_wmma_kernel<2048, 1, 1, 1, 1, 1, 0, 1, 1, 128, 1, false, float, float, true, true>
      <<<1, 256, 0, stream>>>(act2048, Wfc1P, fc1_b, fc1out, Bn);

  // --- head (128 -> 12) + tanh -> d_out (16*12 f32) ---
  head_kernel<<<1, 256, 0, stream>>>(fc1out, head_w, head_b, (float*)d_out);
}
